// HMultiHeadAttention_22239340659157
// MI455X (gfx1250) — matrix operand order repaired and checked by reading, not compiled, not run on hardware
//
#include <hip/hip_runtime.h>
#include <hip/hip_bf16.h>
#include <math.h>

// CDNA5 / gfx1250, wave32. Matrix ops via V_WMMA_F32_16X16X32_BF16.
// Attention K/V tiles staged Global->LDS with global_load_async_to_lds_b128
// (ASYNCcnt), double-buffered per 64-key block.

typedef __attribute__((ext_vector_type(16))) __bf16 v16bf;
typedef __attribute__((ext_vector_type(2)))  __bf16 v2bf;
typedef __attribute__((ext_vector_type(8)))  float  v8f;

#define DEVI static __device__ __forceinline__

DEVI unsigned pkbf(float a, float b) {
  v2bf v; v[0] = (__bf16)a; v[1] = (__bf16)b;
  return __builtin_bit_cast(unsigned, v);
}
DEVI unsigned short bf1(float a) {
  __bf16 h = (__bf16)a;
  return __builtin_bit_cast(unsigned short, h);
}

union BFrag { v16bf v; unsigned u[8]; };

DEVI v8f wmma_bf16(v16bf a, v16bf b, v8f c) {
  return __builtin_amdgcn_wmma_f32_16x16x32_bf16(false, a, false, b, (short)0, c,
                                                 false, false);
}

// Async DMA: global -> LDS, 16 bytes per lane, tracked by ASYNCcnt.
DEVI void async_g2l_b128(unsigned lds_byte_addr, const void* gaddr) {
  asm volatile("global_load_async_to_lds_b128 %0, %1, off"
               :: "v"(lds_byte_addr), "v"((unsigned long long)gaddr)
               : "memory");
}
DEVI void wait_async0() {
  asm volatile("s_wait_asynccnt 0" ::: "memory");
}
DEVI unsigned lds_addr(const void* p) {      // generic ptr low 32 bits == LDS offset
  return (unsigned)(unsigned long long)p;
}

// ---------------------------------------------------------------------------
// Kernel 0: bulk fp32 -> bf16 conversion, 8 elements/thread, 128b loads/stores.
// ---------------------------------------------------------------------------
__global__ __launch_bounds__(256) void cvt_bf16(
    const float* __restrict__ in, unsigned short* __restrict__ out, int n8)
{
  const int i = blockIdx.x * 256 + threadIdx.x;
  if (i >= n8) return;
  const float4* p = (const float4*)in + (size_t)i * 2;
  const float4 a = p[0];
  const float4 b = p[1];
  uint4 o;
  o.x = pkbf(a.x, a.y);
  o.y = pkbf(a.z, a.w);
  o.z = pkbf(b.x, b.y);
  o.w = pkbf(b.z, b.w);
  ((uint4*)out)[i] = o;
}

// ---------------------------------------------------------------------------
// Kernel 1: z = X @ W^T + b (bf16 in, fp32 accum), per-head Lorentz projection.
// mode 0: Q -> [BH][2048][64]; mode 1: K same + time negated; mode 2: V -> [BH][64][2048].
// Wave tile: 16 rows x 64 cols (one head). 4096 waves.
// ---------------------------------------------------------------------------
__global__ __launch_bounds__(256) void hlinear_qkv(
    const unsigned short* __restrict__ X, const unsigned short* __restrict__ W,
    const float* __restrict__ bias, unsigned short* __restrict__ out, int mode)
{
  const int lane = threadIdx.x & 31;
  const int wid  = threadIdx.x >> 5;
  const int gw   = blockIdx.x * 8 + wid;
  const int mt   = gw >> 4;
  const int head = gw & 15;
  const int m0   = mt << 4;
  const int j0   = head << 6;
  const int half = lane >> 4;
  const int ln   = lane & 15;

  v8f acc[4];
  #pragma unroll
  for (int nt = 0; nt < 4; ++nt)
    #pragma unroll
    for (int r = 0; r < 8; ++r) acc[nt][r] = 0.0f;

  const unsigned short* xrow = X + (size_t)(m0 + ln) * 1024;

  for (int k0 = 0; k0 < 1024; k0 += 32) {
    BFrag a;                                          // A: 16x32, row = ln
    #pragma unroll
    for (int p = 0; p < 8; ++p) {
      const int kk = k0 + ((p >> 2) << 4) + (half << 3) + ((p & 3) << 1);
      a.u[p] = *(const unsigned*)(xrow + kk);
    }
    BFrag bm[4];                                      // load ALL B tiles first
    #pragma unroll
    for (int nt = 0; nt < 4; ++nt) {
      const unsigned short* wrow = W + (size_t)(j0 + (nt << 4) + ln) * 1024;
      #pragma unroll
      for (int p = 0; p < 8; ++p)
        bm[nt].u[p] = *(const unsigned*)(wrow + k0 + ((p >> 2) << 4) + (half << 3) + ((p & 3) << 1));
    }
    #pragma unroll
    for (int nt = 0; nt < 4; ++nt)                    // then 4 back-to-back WMMAs
      acc[nt] = wmma_bf16(a.v, bm[nt].v, acc[nt]);
  }

  #pragma unroll
  for (int nt = 0; nt < 4; ++nt) {
    const float bv = bias[j0 + (nt << 4) + ln];
    #pragma unroll
    for (int r = 0; r < 8; ++r) acc[nt][r] += bv;
  }

  float timev[8];
  #pragma unroll
  for (int r = 0; r < 8; ++r) {
    float t = 0.f;
    #pragma unroll
    for (int nt = 0; nt < 4; ++nt) { const float z = acc[nt][r]; t += z * z; }
    #pragma unroll
    for (int m = 1; m < 16; m <<= 1) t += __shfl_xor(t, m, 32);
    const float z0 = __shfl(acc[0][r], half << 4, 32);
    timev[r] = sqrtf(1.0f + t - z0 * z0);             // K = 1
  }
  if (ln == 0) {
    #pragma unroll
    for (int r = 0; r < 8; ++r) acc[0][r] = (mode == 1) ? -timev[r] : timev[r];
  }

  const int b  = m0 >> 11;
  const int l0 = m0 & 2047;
  const size_t bh = (size_t)(b * 16 + head);
  #pragma unroll
  for (int nt = 0; nt < 4; ++nt)
    #pragma unroll
    for (int r = 0; r < 8; ++r) {
      const int row = l0 + r + (half << 3);
      const int c   = (nt << 4) + ln;
      const unsigned short v = bf1(acc[nt][r]);
      if (mode == 2) out[(bh * 64 + c) * 2048 + row] = v;
      else           out[(bh * 2048 + row) * 64 + c] = v;
    }
}

// ---------------------------------------------------------------------------
// Kernel 2: flash attention. Block = 4 waves, one (b,h), 64 query rows.
// K/V 64-key tiles DMA'd to LDS (async, double buffered), shared by the
// 4 waves; softmax P transposed via per-wave LDS; all matmuls bf16 WMMA.
// ---------------------------------------------------------------------------
__global__ __launch_bounds__(128) void hattn(
    const unsigned short* __restrict__ Qh, const unsigned short* __restrict__ Kh,
    const unsigned short* __restrict__ Vt, unsigned short* __restrict__ Ocat)
{
  __shared__ unsigned short sK[2][64 * 72];   // [key][d], 144B rows (16B aligned)
  __shared__ unsigned short sV[2][64 * 72];   // [d][key]
  __shared__ unsigned short sP[4][16 * 72];   // per-wave P staging

  const int tid  = threadIdx.x;
  const int lane = tid & 31;
  const int wid  = tid >> 5;
  const int bh   = blockIdx.x >> 5;           // 0..31
  const int mg   = blockIdx.x & 31;           // 0..31
  const int m0   = (mg * 4 + wid) << 4;       // this wave's 16 query rows
  const int half = lane >> 4;
  const int ln   = lane & 15;

  const unsigned short* Qb = Qh + (size_t)bh * 2048 * 64;
  const unsigned short* Kb = Kh + (size_t)bh * 2048 * 64;
  const unsigned short* Vb = Vt + (size_t)bh * 64 * 2048;

  // Q A-fragments (per wave, K-dim = 64), from global once.
  BFrag qf[2];
  #pragma unroll
  for (int f = 0; f < 2; ++f)
    #pragma unroll
    for (int p = 0; p < 8; ++p) {
      const int kk = (f << 5) + (half << 4) + (p << 1);
      qf[f].u[p] = *(const unsigned*)(Qb + (size_t)(m0 + ln) * 64 + kk);
    }

  v8f o[4];
  float rm[8], rs[8];
  #pragma unroll
  for (int nt = 0; nt < 4; ++nt)
    #pragma unroll
    for (int r = 0; r < 8; ++r) o[nt][r] = 0.0f;
  #pragma unroll
  for (int r = 0; r < 8; ++r) { rm[r] = -3.0e38f; rs[r] = 0.0f; }

  // Cooperative async stage of one 64-key block (K: 64x128B, V: 64x128B).
  // 512 16B-chunks per tensor, 128 threads -> 4 chunks each.
  auto stage = [&](int kb, int buf) {
    const unsigned kbase = lds_addr(&sK[buf][0]);
    const unsigned vbase = lds_addr(&sV[buf][0]);
    #pragma unroll
    for (int i = 0; i < 4; ++i) {
      const int chunk = tid + (i << 7);
      const int row = chunk >> 3;
      const int cc  = chunk & 7;
      async_g2l_b128(kbase + row * 144 + (cc << 4),
                     Kb + (size_t)(kb + row) * 64 + (cc << 3));
      async_g2l_b128(vbase + row * 144 + (cc << 4),
                     Vb + (size_t)row * 2048 + kb + (cc << 3));
    }
  };

  stage(0, 0);
  wait_async0();
  __syncthreads();

  for (int ib = 0; ib < 32; ++ib) {
    const int cur = ib & 1;
    if (ib + 1 < 32) stage((ib + 1) << 6, 1 - cur);

    // ---- S = Q K~^T over this 64-key block (from LDS) ----
    v8f sacc[4];
    #pragma unroll
    for (int nt = 0; nt < 4; ++nt) {
      #pragma unroll
      for (int r = 0; r < 8; ++r) sacc[nt][r] = 0.0f;
      const unsigned short* krow = &sK[cur][((nt << 4) + ln) * 72];
      #pragma unroll
      for (int f = 0; f < 2; ++f) {
        BFrag kf;
        const uint4 q0 = *(const uint4*)(krow + (f << 5) + (half << 4));
        const uint4 q1 = *(const uint4*)(krow + (f << 5) + (half << 4) + 8);
        kf.u[0] = q0.x; kf.u[1] = q0.y; kf.u[2] = q0.z; kf.u[3] = q0.w;
        kf.u[4] = q1.x; kf.u[5] = q1.y; kf.u[6] = q1.z; kf.u[7] = q1.w;
        sacc[nt] = wmma_bf16(qf[f].v, kf.v, sacc[nt]);
      }
      // scores = (2K + 2*ip)/sqrt(64) = 0.25*ip + 0.25
      #pragma unroll
      for (int r = 0; r < 8; ++r) sacc[nt][r] = fmaf(sacc[nt][r], 0.25f, 0.25f);
    }

    // ---- online softmax ----
    float alpha[8];
    #pragma unroll
    for (int r = 0; r < 8; ++r) {
      float t = fmaxf(fmaxf(sacc[0][r], sacc[1][r]), fmaxf(sacc[2][r], sacc[3][r]));
      #pragma unroll
      for (int m = 1; m < 16; m <<= 1) t = fmaxf(t, __shfl_xor(t, m, 32));
      const float nm = fmaxf(rm[r], t);
      alpha[r] = __expf(rm[r] - nm);
      rm[r] = nm;
    }
    #pragma unroll
    for (int nt = 0; nt < 4; ++nt)
      #pragma unroll
      for (int r = 0; r < 8; ++r) sacc[nt][r] = __expf(sacc[nt][r] - rm[r]);
    #pragma unroll
    for (int r = 0; r < 8; ++r) {
      float t = sacc[0][r] + sacc[1][r] + sacc[2][r] + sacc[3][r];
      #pragma unroll
      for (int m = 1; m < 16; m <<= 1) t += __shfl_xor(t, m, 32);
      rs[r] = rs[r] * alpha[r] + t;
    }
    #pragma unroll
    for (int nt = 0; nt < 4; ++nt)
      #pragma unroll
      for (int r = 0; r < 8; ++r) o[nt][r] *= alpha[r];

    // ---- P: C-layout -> per-wave LDS -> A-fragments ----
    #pragma unroll
    for (int nt = 0; nt < 4; ++nt)
      #pragma unroll
      for (int r = 0; r < 8; ++r)
        sP[wid][(r + (half << 3)) * 72 + (nt << 4) + ln] = bf1(sacc[nt][r]);

    BFrag pf[2];
    #pragma unroll
    for (int f = 0; f < 2; ++f) {
      const unsigned short* pp = &sP[wid][ln * 72 + (f << 5) + (half << 4)];
      const uint4 q0 = *(const uint4*)pp;
      const uint4 q1 = *(const uint4*)(pp + 8);
      pf[f].u[0] = q0.x; pf[f].u[1] = q0.y; pf[f].u[2] = q0.z; pf[f].u[3] = q0.w;
      pf[f].u[4] = q1.x; pf[f].u[5] = q1.y; pf[f].u[6] = q1.z; pf[f].u[7] = q1.w;
    }

    // ---- O += P V (from LDS) ----
    #pragma unroll
    for (int nt = 0; nt < 4; ++nt) {
      const unsigned short* vrow = &sV[cur][((nt << 4) + ln) * 72];
      #pragma unroll
      for (int f = 0; f < 2; ++f) {
        BFrag vf;
        const uint4 q0 = *(const uint4*)(vrow + (f << 5) + (half << 4));
        const uint4 q1 = *(const uint4*)(vrow + (f << 5) + (half << 4) + 8);
        vf.u[0] = q0.x; vf.u[1] = q0.y; vf.u[2] = q0.z; vf.u[3] = q0.w;
        vf.u[4] = q1.x; vf.u[5] = q1.y; vf.u[6] = q1.z; vf.u[7] = q1.w;
        o[nt] = wmma_bf16(pf[f].v, vf.v, o[nt]);
      }
    }

    wait_async0();        // own DMA for block ib+1 done
    __syncthreads();      // everyone done computing ib & staging ib+1
  }

  // mu = O / rowsum; Lorentz-normalize over the head's 64 dims
  #pragma unroll
  for (int r = 0; r < 8; ++r) {
    const float inv = 1.0f / rs[r];
    #pragma unroll
    for (int nt = 0; nt < 4; ++nt) o[nt][r] *= inv;
  }
  float scl[8];
  #pragma unroll
  for (int r = 0; r < 8; ++r) {
    float t = 0.f;
    #pragma unroll
    for (int nt = 0; nt < 4; ++nt) { const float z = o[nt][r]; t += z * z; }
    #pragma unroll
    for (int m = 1; m < 16; m <<= 1) t += __shfl_xor(t, m, 32);
    const float mu0   = __shfl(o[0][r], half << 4, 32);
    const float inner = t - 2.0f * mu0 * mu0;
    scl[r] = 1.0f / sqrtf(fmaxf(fabsf(inner), 1e-6f));
  }
  const int b = bh >> 4, h = bh & 15;
  const size_t rowbase = (size_t)b * 2048 + m0;
  #pragma unroll
  for (int nt = 0; nt < 4; ++nt)
    #pragma unroll
    for (int r = 0; r < 8; ++r)
      Ocat[(rowbase + r + (half << 3)) * 1024 + (h << 6) + (nt << 4) + ln] =
          bf1(o[nt][r] * scl[r]);
}

// ---------------------------------------------------------------------------
// Kernel 3: Z = cat @ W_O^T + b -> fp32 d_out; deterministic per-(row,coltile)
// partial sum of squares (excluding global column 0).
// ---------------------------------------------------------------------------
__global__ __launch_bounds__(256) void hout_proj(
    const unsigned short* __restrict__ A, const unsigned short* __restrict__ W,
    const float* __restrict__ bias, float* __restrict__ out,
    float* __restrict__ partials)
{
  const int lane = threadIdx.x & 31;
  const int wid  = threadIdx.x >> 5;
  const int gw   = blockIdx.x * 8 + wid;
  const int mt   = gw >> 4;
  const int ct   = gw & 15;
  const int m0   = mt << 4;
  const int j0   = ct << 6;
  const int half = lane >> 4;
  const int ln   = lane & 15;

  v8f acc[4];
  #pragma unroll
  for (int nt = 0; nt < 4; ++nt)
    #pragma unroll
    for (int r = 0; r < 8; ++r) acc[nt][r] = 0.0f;

  const unsigned short* arow = A + (size_t)(m0 + ln) * 1024;

  for (int k0 = 0; k0 < 1024; k0 += 32) {
    BFrag a;
    #pragma unroll
    for (int p = 0; p < 8; ++p) {
      const int kk = k0 + ((p >> 2) << 4) + (half << 3) + ((p & 3) << 1);
      a.u[p] = *(const unsigned*)(arow + kk);
    }
    BFrag bm[4];
    #pragma unroll
    for (int nt = 0; nt < 4; ++nt) {
      const unsigned short* wrow = W + (size_t)(j0 + (nt << 4) + ln) * 1024;
      #pragma unroll
      for (int p = 0; p < 8; ++p)
        bm[nt].u[p] = *(const unsigned*)(wrow + k0 + ((p >> 2) << 4) + (half << 3) + ((p & 3) << 1));
    }
    #pragma unroll
    for (int nt = 0; nt < 4; ++nt)
      acc[nt] = wmma_bf16(a.v, bm[nt].v, acc[nt]);
  }

  #pragma unroll
  for (int nt = 0; nt < 4; ++nt) {
    const float bv = bias[j0 + (nt << 4) + ln];
    #pragma unroll
    for (int r = 0; r < 8; ++r) acc[nt][r] += bv;
  }

  float ssq[8];
  #pragma unroll
  for (int r = 0; r < 8; ++r) {
    float t = 0.f;
    #pragma unroll
    for (int nt = 0; nt < 4; ++nt) {
      const float z = acc[nt][r];
      const bool excl = (j0 == 0) && (nt == 0) && (ln == 0);  // global col 0
      t += excl ? 0.f : z * z;
    }
    #pragma unroll
    for (int m = 1; m < 16; m <<= 1) t += __shfl_xor(t, m, 32);
    ssq[r] = t;
  }

  #pragma unroll
  for (int nt = 0; nt < 4; ++nt)
    #pragma unroll
    for (int r = 0; r < 8; ++r)
      out[(size_t)(m0 + r + (half << 3)) * 1024 + j0 + (nt << 4) + ln] = acc[nt][r];

  #pragma unroll
  for (int r = 0; r < 8; ++r)
    if (ln == r)
      partials[(size_t)(m0 + r + (half << 3)) * 16 + ct] = ssq[r];
}

// Kernel 4: finalize time column of the full-width projection.
__global__ void hfinal(float* __restrict__ out, const float* __restrict__ partials) {
  const int n = blockIdx.x * 256 + threadIdx.x;
  if (n >= 4096) return;
  float s = 0.f;
  #pragma unroll
  for (int t = 0; t < 16; ++t) s += partials[n * 16 + t];
  out[(size_t)n * 1024] = sqrtf(1.0f + s);
}

extern "C" void kernel_launch(void* const* d_in, const int* in_sizes, int n_in,
                              void* d_out, int out_size, void* d_ws, size_t ws_size,
                              hipStream_t stream) {
  (void)in_sizes; (void)n_in; (void)out_size; (void)ws_size;
  const float* query = (const float*)d_in[0];
  const float* key   = (const float*)d_in[1];
  const float* value = (const float*)d_in[2];
  const float* W_Q   = (const float*)d_in[3];
  const float* b_Q   = (const float*)d_in[4];
  const float* W_K   = (const float*)d_in[5];
  const float* b_K   = (const float*)d_in[6];
  const float* W_V   = (const float*)d_in[7];
  const float* b_V   = (const float*)d_in[8];
  const float* W_O   = (const float*)d_in[9];
  const float* b_O   = (const float*)d_in[10];

  char* ws = (char*)d_ws;
  unsigned short* Qh   = (unsigned short*)(ws);                        //  8 MB
  unsigned short* Kh   = (unsigned short*)(ws + (size_t)(8u  << 20));  //  8 MB
  unsigned short* Vt   = (unsigned short*)(ws + (size_t)(16u << 20));  //  8 MB
  unsigned short* Ocat = (unsigned short*)(ws + (size_t)(24u << 20));  //  8 MB
  unsigned short* Xbf  = (unsigned short*)(ws + (size_t)(32u << 20));  //  8 MB staging
  unsigned short* Wbf  = (unsigned short*)(ws + (size_t)(40u << 20));  //  2 MB staging
  float*          part = (float*)(ws + (size_t)(42u << 20));           // 256 KB

  const int NX8 = (2 * 2048 * 1024) / 8;
  const int NW8 = (1024 * 1024) / 8;

  cvt_bf16<<<NX8 / 256, 256, 0, stream>>>(query, Xbf, NX8);
  cvt_bf16<<<NW8 / 256, 256, 0, stream>>>(W_Q, Wbf, NW8);
  hlinear_qkv<<<512, 256, 0, stream>>>(Xbf, Wbf, b_Q, Qh, 0);

  cvt_bf16<<<NX8 / 256, 256, 0, stream>>>(key, Xbf, NX8);
  cvt_bf16<<<NW8 / 256, 256, 0, stream>>>(W_K, Wbf, NW8);
  hlinear_qkv<<<512, 256, 0, stream>>>(Xbf, Wbf, b_K, Kh, 1);

  cvt_bf16<<<NX8 / 256, 256, 0, stream>>>(value, Xbf, NX8);
  cvt_bf16<<<NW8 / 256, 256, 0, stream>>>(W_V, Wbf, NW8);
  hlinear_qkv<<<512, 256, 0, stream>>>(Xbf, Wbf, b_V, Vt, 2);

  hattn<<<1024, 128, 0, stream>>>(Qh, Kh, Vt, Ocat);

  cvt_bf16<<<NW8 / 256, 256, 0, stream>>>(W_O, Wbf, NW8);
  hout_proj<<<512, 256, 0, stream>>>(Ocat, Wbf, b_O, (float*)d_out, part);
  hfinal<<<16, 256, 0, stream>>>((float*)d_out, part);
}
